// BiRNNDecoder_38190849196288
// MI455X (gfx1250) — matrix operand order repaired and checked by reading, not compiled, not run on hardware
//
#include <hip/hip_runtime.h>

typedef __attribute__((ext_vector_type(16))) _Float16 v16h;
typedef __attribute__((ext_vector_type(8)))  float    v8f;
typedef __attribute__((ext_vector_type(8)))  int      v8i;
typedef __attribute__((ext_vector_type(2)))  _Float16 h2;

__device__ __forceinline__ float fast_sigmoid(float v) {
    float e = __builtin_amdgcn_exp2f(-1.442695040888963f * v);
    return __builtin_amdgcn_rcpf(1.0f + e);
}

__device__ __forceinline__ float fast_tanh(float v) {
    v = fminf(9.0f, fmaxf(-9.0f, v));
    float e = __builtin_amdgcn_exp2f(2.885390081777927f * v);   // exp(2v)
    return (e - 1.0f) * __builtin_amdgcn_rcpf(e + 1.0f);
}

// Exchange a dword between lane l and lane l^16 (SWAPX16: and=0x1f, or=0, xor=0x10)
__device__ __forceinline__ int swap16(int v) {
    return __builtin_amdgcn_ds_swizzle(v, 0x401F);
}

// One wave = one 16-row batch tile of one direction. Runs the full T-step GRU
// scan with h resident in registers; 6x v_wmma_f32_16x16x32_f16 per step
// computing gh^T = W_hh @ h^T (A = W blocks, loop-invariant in registers).
__global__ __launch_bounds__(32)
void gru_bidir_scan(const float* __restrict__ x,
                    const float* __restrict__ wih_f, const float* __restrict__ whh_f,
                    const float* __restrict__ bih_f, const float* __restrict__ bhh_f,
                    const float* __restrict__ wih_b, const float* __restrict__ whh_b,
                    const float* __restrict__ bih_b, const float* __restrict__ bhh_b,
                    float* __restrict__ hstage, int T)
{
    const int  lane = threadIdx.x;
    const int  lo   = lane & 15;          // batch column within tile (N of WMMA)
    const int  hi   = lane >> 4;          // half-wave select
    const int  dir  = blockIdx.x & 1;
    const long rowbase = (long)(blockIdx.x >> 1) * 16;

    const float* wih = dir ? wih_b : wih_f;
    const float* whh = dir ? whh_b : whh_f;
    const float* bih = dir ? bih_b : bih_f;
    const float* bhh = dir ? bhh_b : bhh_f;

    // ---- Preload W_hh as six 16x32 f16 A-matrices (loop-invariant) ----------
    // A order used here (contiguous K, B matches): lanes 0-15 row M=lane: halves 0..15 -> K=0..15;
    // lanes 16-31 row M=lane-16: halves 0..15 -> K=16..31.
    v16h A[6];
#pragma unroll
    for (int tl = 0; tl < 6; ++tl) {
        const float* wr = whh + (tl * 16 + lo) * 32;
#pragma unroll
        for (int p = 0; p < 8; ++p) {
            A[tl][p]     = (_Float16)wr[16 * hi + p];
            A[tl][p + 8] = (_Float16)wr[16 * hi + 8 + p];
        }
    }

    // ---- Per-lane gate constants in D-tile layout: gate g = tl*16 + k + 8*hi.
    // r/z gates: fold b_ih+b_hh into one bias. n gate: b_ih only here; b_hh_n
    // must live inside gh_n (it is multiplied by r), so it goes into WMMA C.
    float wihc[6][8], biasc[6][8], bhn[2][8];
#pragma unroll
    for (int tl = 0; tl < 6; ++tl) {
#pragma unroll
        for (int k = 0; k < 8; ++k) {
            int g = tl * 16 + k + 8 * hi;
            wihc[tl][k]  = wih[g];
            biasc[tl][k] = bih[g] + (tl < 4 ? bhh[g] : 0.0f);
        }
    }
#pragma unroll
    for (int k = 0; k < 8; ++k) {
        bhn[0][k] = bhh[64 + k + 8 * hi];
        bhn[1][k] = bhh[80 + k + 8 * hi];
    }

    // h^T kept in C/D layout: H0 holds j = k+8*hi (0..15), H1 holds j+16.
    float H0[8], H1[8];
#pragma unroll
    for (int k = 0; k < 8; ++k) { H0[k] = 0.0f; H1[k] = 0.0f; }

    v8i bb;
#pragma unroll
    for (int v = 0; v < 8; ++v) bb[v] = 0;
    v16h Bm = __builtin_bit_cast(v16h, bb);   // h^T as B operand (zero at t=0)

    const float* xr = x + (rowbase + lo) * (long)T;

    for (int t = 0; t < T; ++t) {
        const int   tt = dir ? (T - 1 - t) : t;
        const float xv = xr[tt];

        // gh^T = W_hh @ h^T  (+ b_hh_n in C for the n tiles)
        v8f c0 = {}, c1 = {}, c2 = {}, c3 = {}, c4, c5;
#pragma unroll
        for (int k = 0; k < 8; ++k) { c4[k] = bhn[0][k]; c5[k] = bhn[1][k]; }

        c0 = __builtin_amdgcn_wmma_f32_16x16x32_f16(false, A[0], false, Bm, (short)0, c0, false, false);
        c1 = __builtin_amdgcn_wmma_f32_16x16x32_f16(false, A[1], false, Bm, (short)0, c1, false, false);
        c2 = __builtin_amdgcn_wmma_f32_16x16x32_f16(false, A[2], false, Bm, (short)0, c2, false, false);
        c3 = __builtin_amdgcn_wmma_f32_16x16x32_f16(false, A[3], false, Bm, (short)0, c3, false, false);
        c4 = __builtin_amdgcn_wmma_f32_16x16x32_f16(false, A[4], false, Bm, (short)0, c4, false, false);
        c5 = __builtin_amdgcn_wmma_f32_16x16x32_f16(false, A[5], false, Bm, (short)0, c5, false, false);

        // Gate math; co-executes with WMMA pipe (exp/rcp are TRANS ops).
#pragma unroll
        for (int k = 0; k < 8; ++k) {
            {
                float r = fast_sigmoid(c0[k] + xv * wihc[0][k] + biasc[0][k]);
                float z = fast_sigmoid(c2[k] + xv * wihc[2][k] + biasc[2][k]);
                float n = fast_tanh(xv * wihc[4][k] + biasc[4][k] + r * c4[k]);
                H0[k] = n + z * (H0[k] - n);
            }
            {
                float r = fast_sigmoid(c1[k] + xv * wihc[1][k] + biasc[1][k]);
                float z = fast_sigmoid(c3[k] + xv * wihc[3][k] + biasc[3][k]);
                float n = fast_tanh(xv * wihc[5][k] + biasc[5][k] + r * c5[k]);
                H1[k] = n + z * (H1[k] - n);
            }
        }

        // Repack h^T into B layout for the next step, fully in-register:
        // pack f32->f16 pairs (K-adjacent = in-lane), swap half-waves, select.
        int ph0[4], ph1[4];
#pragma unroll
        for (int v = 0; v < 4; ++v) {
            ph0[v] = __builtin_bit_cast(int, __builtin_amdgcn_cvt_pkrtz(H0[2 * v], H0[2 * v + 1]));
            ph1[v] = __builtin_bit_cast(int, __builtin_amdgcn_cvt_pkrtz(H1[2 * v], H1[2 * v + 1]));
        }
#pragma unroll
        for (int v = 0; v < 4; ++v) {
            int sw0 = swap16(ph0[v]);
            int sw1 = swap16(ph1[v]);
            bb[v]     = hi ? sw1 : ph0[v];   // K = 16*hi + 2v, 2v+1
            bb[v + 4] = hi ? ph1[v] : sw0;   // K = 16*hi + 8 + 2v, 2v+1
        }
        Bm = __builtin_bit_cast(v16h, bb);
    }

    // Stage pooled hidden state: hstage[b][dir*32 + j], j = column of h.
    float* hs = hstage + (rowbase + lo) * 64 + dir * 32;
#pragma unroll
    for (int k = 0; k < 8; ++k) {
        hs[k + 8 * hi]      = H0[k];
        hs[16 + k + 8 * hi] = H1[k];
    }
}

// out[B,256] = pooled[B,64] @ w_out^T + b_out, one wave per 16-row batch tile,
// two accumulating WMMAs (K=64) per 16x16 output tile.
__global__ __launch_bounds__(32)
void out_proj(const float* __restrict__ hstage,
              const float* __restrict__ w_out,
              const float* __restrict__ b_out,
              float* __restrict__ out)
{
    const int  lane = threadIdx.x;
    const int  lo   = lane & 15;
    const int  hi   = lane >> 4;
    const long rowbase = (long)blockIdx.x * 16;

    // A = pooled rows, split into K-chunks [0..31] and [32..63].
    v16h A0, A1;
    const float* hr = hstage + (rowbase + lo) * 64;
#pragma unroll
    for (int p = 0; p < 8; ++p) {
        A0[p]     = (_Float16)hr[16 * hi + p];
        A0[p + 8] = (_Float16)hr[16 * hi + 8 + p];
        A1[p]     = (_Float16)hr[32 + 16 * hi + p];
        A1[p + 8] = (_Float16)hr[40 + 16 * hi + p];
    }

    for (int nt = 0; nt < 16; ++nt) {
        // B element (K=c, N=n) = w_out[n*64 + c]; lane column n = nt*16+lo,
        // halves are 16 contiguous c values starting at 32*chunk + 16*hi.
        const float* wr = w_out + (long)(nt * 16 + lo) * 64 + 16 * hi;
        v16h B0, B1;
#pragma unroll
        for (int j = 0; j < 16; ++j) {
            B0[j] = (_Float16)wr[j];
            B1[j] = (_Float16)wr[32 + j];
        }
        const float bo = b_out[nt * 16 + lo];
        v8f c;
#pragma unroll
        for (int k = 0; k < 8; ++k) c[k] = bo;

        c = __builtin_amdgcn_wmma_f32_16x16x32_f16(false, A0, false, B0, (short)0, c, false, false);
        c = __builtin_amdgcn_wmma_f32_16x16x32_f16(false, A1, false, B1, (short)0, c, false, false);

#pragma unroll
        for (int k = 0; k < 8; ++k)
            out[(rowbase + k + 8 * hi) * 256 + nt * 16 + lo] = c[k];
    }
}

extern "C" void kernel_launch(void* const* d_in, const int* in_sizes, int n_in,
                              void* d_out, int out_size, void* d_ws, size_t ws_size,
                              hipStream_t stream) {
    (void)n_in; (void)ws_size;
    const float* x     = (const float*)d_in[0];
    const float* wih_f = (const float*)d_in[1];
    const float* whh_f = (const float*)d_in[2];
    const float* bih_f = (const float*)d_in[3];
    const float* bhh_f = (const float*)d_in[4];
    const float* wih_b = (const float*)d_in[5];
    const float* whh_b = (const float*)d_in[6];
    const float* bih_b = (const float*)d_in[7];
    const float* bhh_b = (const float*)d_in[8];
    const float* w_out = (const float*)d_in[9];
    const float* b_out = (const float*)d_in[10];

    const int B = out_size / 256;          // 8192
    const int T = in_sizes[0] / B;         // 524 (D == 1)
    float* hstage = (float*)d_ws;          // B*64 floats = 2 MB

    gru_bidir_scan<<<dim3(2 * (B / 16)), dim3(32), 0, stream>>>(
        x, wih_f, whh_f, bih_f, bhh_f, wih_b, whh_b, bih_b, bhh_b, hstage, T);
    out_proj<<<dim3(B / 16), dim3(32), 0, stream>>>(hstage, w_out, b_out, (float*)d_out);
}
